// GAT_35450660061441
// MI455X (gfx1250) — compile-verified, matrix operand order checked by reading, not run on hardware
//
#include <hip/hip_runtime.h>
#include <hip/hip_bf16.h>

typedef __bf16 bf16;
typedef __attribute__((ext_vector_type(16))) __bf16 v16bf;
typedef __attribute__((ext_vector_type(8)))  __bf16 v8bf;
typedef __attribute__((ext_vector_type(8)))  float   v8f;

#define NNODES 50000
#define FIN    128
#define FOUT   64
#define NEG_SLOPE 0.2f

// ---------------------------------------------------------------------------
// f32 -> bf16 conversion (activations)
// ---------------------------------------------------------------------------
__global__ void gat_cvt_bf16(const float* __restrict__ src, bf16* __restrict__ dst, int n) {
    int t = blockIdx.x * blockDim.x + threadIdx.x;
    if (t < n) dst[t] = (bf16)src[t];
}

// ---------------------------------------------------------------------------
// zero fill
// ---------------------------------------------------------------------------
__global__ void gat_zero_f32(float* __restrict__ p, int n) {
    int t = blockIdx.x * blockDim.x + threadIdx.x;
    if (t < n) p[t] = 0.0f;
}

// ---------------------------------------------------------------------------
// Pack W [128, Nw] f32 into fragment-swizzled bf16 so each lane's B fragment
// is a contiguous 32B chunk:
//   out[((tn*4 + s)*32 + lane)*16 + i] = (bf16) W[k * Nw + tn*16 + (lane&15)]
//   k = s*32 + (i<8 ? 8*half + i : 16 + 8*half + (i-8)),  half = lane>>4
// ---------------------------------------------------------------------------
__global__ void gat_pack_w(const float* __restrict__ W, bf16* __restrict__ out, int Nw) {
    int t = blockIdx.x * blockDim.x + threadIdx.x;
    if (t >= Nw * 128) return;
    int i    = t & 15;
    int lane = (t >> 4) & 31;
    int s    = (t >> 9) & 3;
    int tn   = t >> 11;
    int half = lane >> 4;
    int col  = tn * 16 + (lane & 15);
    int k    = s * 32 + (i < 8 ? half * 8 + i : 16 + half * 8 + (i - 8));
    out[t] = (bf16)W[k * Nw + col];
}

// ---------------------------------------------------------------------------
// GEMM: C[M,Nw] = A[M,128](bf16) x B[128,Nw](bf16, pre-swizzled), f32 accum.
// One wave32 per 16x64 strip: A fragment loaded once per k-step, 4 WMMAs.
// A layout per ISA: half = lane>>4, M = tm*16 + (lane&15),
//   element i -> K = 8*half + i (i<8), 16 + 8*half + (i-8) (i>=8).
// C/D: vgpr r -> M = r + 8*half, N = lane&15.
// ---------------------------------------------------------------------------
__global__ void gat_gemm_wmma(const bf16* __restrict__ A, const bf16* __restrict__ Bs,
                              float* __restrict__ C, int M, int Nw) {
    const int K = 128;
    int lane = threadIdx.x & 31;
    int wave = blockIdx.x * (blockDim.x >> 5) + (threadIdx.x >> 5);
    int ngrp = Nw >> 6;                 // groups of 4 n-tiles
    int tm   = wave / ngrp;
    int tg   = wave % ngrp;
    if (tm * 16 >= M) return;

    int half = lane >> 4;
    int mrow = tm * 16 + (lane & 15);

    v8f a0 = {}, a1 = {}, a2 = {}, a3 = {};
    #pragma unroll
    for (int s = 0; s < 4; ++s) {
        const bf16* arow = A + (size_t)mrow * K + s * 32 + half * 8;
        v8bf alo = *(const v8bf*)(arow);
        v8bf ahi = *(const v8bf*)(arow + 16);
        v16bf af = __builtin_shufflevector(alo, ahi,
                                           0,1,2,3,4,5,6,7,8,9,10,11,12,13,14,15);
        const bf16* bbase = Bs + ((size_t)(tg * 4) * 4 + s) * 512 + lane * 16;
        v16bf b0 = *(const v16bf*)(bbase + 0 * 2048);
        a0 = __builtin_amdgcn_wmma_f32_16x16x32_bf16(false, af, false, b0, (short)0, a0, false, false);
        v16bf b1 = *(const v16bf*)(bbase + 1 * 2048);
        a1 = __builtin_amdgcn_wmma_f32_16x16x32_bf16(false, af, false, b1, (short)0, a1, false, false);
        v16bf b2 = *(const v16bf*)(bbase + 2 * 2048);
        a2 = __builtin_amdgcn_wmma_f32_16x16x32_bf16(false, af, false, b2, (short)0, a2, false, false);
        v16bf b3 = *(const v16bf*)(bbase + 3 * 2048);
        a3 = __builtin_amdgcn_wmma_f32_16x16x32_bf16(false, af, false, b3, (short)0, a3, false, false);
    }
    #pragma unroll
    for (int r = 0; r < 8; ++r) {
        size_t row = (size_t)(tm * 16 + r + 8 * half) * Nw + (lane & 15);
        int nb = tg * 64;
        C[row + nb +  0] = a0[r];
        C[row + nb + 16] = a1[r];
        C[row + nb + 32] = a2[r];
        C[row + nb + 48] = a3[r];
    }
}

// ---------------------------------------------------------------------------
// Per (node, head): attention scalars s_src/s_dst, init max/denom.
// ---------------------------------------------------------------------------
__global__ void gat_node_prep(const float* __restrict__ h,
                              const float* __restrict__ asrc,
                              const float* __restrict__ adst,
                              float* __restrict__ ss, float* __restrict__ sd,
                              float* __restrict__ mx, float* __restrict__ dn,
                              int n_nodes, int H, int C) {
    int t = blockIdx.x * blockDim.x + threadIdx.x;
    if (t >= n_nodes * H) return;
    int n  = t / H;
    int hd = t % H;
    const float* hp = h + (size_t)n * H * C + hd * C;
    const float* as = asrc + hd * C;
    const float* ad = adst + hd * C;
    float s1 = 0.f, s2 = 0.f;
    for (int c = 0; c < C; ++c) {
        float v = hp[c];
        s1 += v * as[c];
        s2 += v * ad[c];
    }
    ss[t] = s1;
    sd[t] = s2;
    mx[t] = -1e30f;
    dn[t] = 0.f;
}

__device__ __forceinline__ float gat_lrelu(float x) {
    return x > 0.f ? x : NEG_SLOPE * x;
}

// Order-preserving float atomic max via int/uint atomics.
__device__ __forceinline__ void gat_atomic_max_f32(float* addr, float val) {
    if (val >= 0.f)
        atomicMax((int*)addr, __float_as_int(val));
    else
        atomicMin((unsigned int*)addr, (unsigned int)__float_as_int(val));
}

// ---------------------------------------------------------------------------
// Pass 1: segment max over incoming edges per (dst, head).
// ---------------------------------------------------------------------------
template <int H>
__global__ void gat_edge_max(const int* __restrict__ src, const int* __restrict__ dst,
                             const float* __restrict__ ss, const float* __restrict__ sd,
                             float* __restrict__ mx, int E) {
    int e = blockIdx.x * blockDim.x + threadIdx.x;
    if (e >= E) return;
    int s = src[e], d = dst[e];
    #pragma unroll
    for (int hd = 0; hd < H; ++hd) {
        float ee = gat_lrelu(ss[s * H + hd] + sd[d * H + hd]);
        gat_atomic_max_f32(&mx[d * H + hd], ee);
    }
}

// ---------------------------------------------------------------------------
// Pass 2: denom = segment_sum(exp(e - m)).
// ---------------------------------------------------------------------------
template <int H>
__global__ void gat_edge_expsum(const int* __restrict__ src, const int* __restrict__ dst,
                                const float* __restrict__ ss, const float* __restrict__ sd,
                                const float* __restrict__ mx, float* __restrict__ dn,
                                int E) {
    int e = blockIdx.x * blockDim.x + threadIdx.x;
    if (e >= E) return;
    int s = src[e], d = dst[e];
    #pragma unroll
    for (int hd = 0; hd < H; ++hd) {
        float ee = gat_lrelu(ss[s * H + hd] + sd[d * H + hd]);
        atomicAdd(&dn[d * H + hd], __expf(ee - mx[d * H + hd]));
    }
}

// ---------------------------------------------------------------------------
// Pass 3: out[dst] += alpha * h[src]. One wave32 per edge; lanes stride
// channels -> fully coalesced 128B loads / atomic adds per iteration.
// ---------------------------------------------------------------------------
template <int H, int C>
__global__ void gat_edge_aggr(const int* __restrict__ src, const int* __restrict__ dst,
                              const float* __restrict__ ss, const float* __restrict__ sd,
                              const float* __restrict__ mx, const float* __restrict__ dn,
                              const float* __restrict__ h, float* __restrict__ out,
                              int E) {
    int lane = threadIdx.x & 31;
    int e = blockIdx.x * (blockDim.x >> 5) + (threadIdx.x >> 5);
    if (e >= E) return;
    int s = src[e], d = dst[e];

    float alpha[H];
    #pragma unroll
    for (int hd = 0; hd < H; ++hd) {
        float ee = gat_lrelu(ss[s * H + hd] + sd[d * H + hd]);
        alpha[hd] = __expf(ee - mx[d * H + hd]) / (dn[d * H + hd] + 1e-16f);
    }
    #pragma unroll
    for (int k = 0; k < (H * C) / 32; ++k) {
        int chan = k * 32 + lane;
        const int hd = (k * 32) / C;   // compile-time (C multiple of 32)
        atomicAdd(&out[(size_t)d * H * C + chan], h[(size_t)s * H * C + chan] * alpha[hd]);
    }
}

// ---------------------------------------------------------------------------
// Finalize: add bias; hidden layers apply ELU and emit bf16 activations for
// the next layer's WMMA GEMM; last layer writes f32 output.
// ---------------------------------------------------------------------------
__global__ void gat_finalize(const float* __restrict__ acc, const float* __restrict__ bias,
                             bf16* __restrict__ xb, float* __restrict__ outF,
                             int n_nodes, int Hw, int apply_elu) {
    int t = blockIdx.x * blockDim.x + threadIdx.x;
    if (t >= n_nodes * Hw) return;
    int c = t % Hw;
    float v = acc[t] + bias[c];
    if (apply_elu) {
        v = v > 0.f ? v : (__expf(v) - 1.0f);
        xb[t] = (bf16)v;
    } else {
        outF[t] = v;
    }
}

// ---------------------------------------------------------------------------
extern "C" void kernel_launch(void* const* d_in, const int* in_sizes, int n_in,
                              void* d_out, int out_size, void* d_ws, size_t ws_size,
                              hipStream_t stream) {
    const float* x      = (const float*)d_in[0];
    const int*   eidx   = (const int*)d_in[1];
    const float* Wstk   = (const float*)d_in[2];   // [4,128,128]
    const float* asrcS  = (const float*)d_in[3];   // [4,4,32]
    const float* adstS  = (const float*)d_in[4];
    const float* bS     = (const float*)d_in[5];   // [4,128]
    const float* Wlast  = (const float*)d_in[6];   // [128,64]
    const float* asrcL  = (const float*)d_in[7];   // [1,64]
    const float* adstL  = (const float*)d_in[8];
    const float* bL     = (const float*)d_in[9];   // [64]

    const int N = NNODES;
    const int E = in_sizes[1] / 2;
    const int* src = eidx;
    const int* dst = eidx + E;

    // workspace carve (256B aligned)
    char* ws = (char*)d_ws;
    auto carve = [&](size_t bytes) -> void* {
        void* p = (void*)ws;
        ws += (bytes + 255) & ~(size_t)255;
        return p;
    };
    bf16*  xb  = (bf16*) carve((size_t)N * FIN * sizeof(bf16));            // activations (bf16)
    bf16*  Wb  = (bf16*) carve((size_t)(4 * 16384 + 8192) * sizeof(bf16)); // swizzled weights
    float* h   = (float*)carve((size_t)N * FIN * sizeof(float));           // post-GEMM features
    float* ssb = (float*)carve((size_t)N * 4 * sizeof(float));
    float* sdb = (float*)carve((size_t)N * 4 * sizeof(float));
    float* mxb = (float*)carve((size_t)N * 4 * sizeof(float));
    float* dnb = (float*)carve((size_t)N * 4 * sizeof(float));
    float* acc = (float*)carve((size_t)N * FIN * sizeof(float));

    const int TB = 256;
    auto cdiv = [](int a, int b) { return (a + b - 1) / b; };

    // Input features to bf16; weights to fragment-swizzled bf16.
    gat_cvt_bf16<<<cdiv(N * FIN, TB), TB, 0, stream>>>(x, xb, N * FIN);
    for (int l = 0; l < 4; ++l)
        gat_pack_w<<<cdiv(16384, TB), TB, 0, stream>>>(Wstk + l * 16384, Wb + l * 16384, 128);
    gat_pack_w<<<cdiv(8192, TB), TB, 0, stream>>>(Wlast, Wb + 4 * 16384, 64);

    // -------- layers 0..3 : 128 -> 128, H=4, C=32, ELU --------
    for (int l = 0; l < 4; ++l) {
        const int Nw = FIN, H = 4, C = 32;
        int waves = (N / 16) * (Nw / 64);
        gat_gemm_wmma<<<cdiv(waves, TB / 32), TB, 0, stream>>>(xb, Wb + l * 16384, h, N, Nw);
        gat_node_prep<<<cdiv(N * H, TB), TB, 0, stream>>>(
            h, asrcS + l * H * C, adstS + l * H * C, ssb, sdb, mxb, dnb, N, H, C);
        gat_zero_f32<<<cdiv(N * Nw, TB), TB, 0, stream>>>(acc, N * Nw);
        gat_edge_max<4><<<cdiv(E, TB), TB, 0, stream>>>(src, dst, ssb, sdb, mxb, E);
        gat_edge_expsum<4><<<cdiv(E, TB), TB, 0, stream>>>(src, dst, ssb, sdb, mxb, dnb, E);
        gat_edge_aggr<4, 32><<<cdiv(E, TB / 32), TB, 0, stream>>>(
            src, dst, ssb, sdb, mxb, dnb, h, acc, E);
        gat_finalize<<<cdiv(N * Nw, TB), TB, 0, stream>>>(
            acc, bS + l * Nw, xb, nullptr, N, Nw, 1);
    }

    // -------- last layer : 128 -> 64, H=1, C=64, no activation --------
    {
        const int Nw = FOUT, H = 1, C = FOUT;
        int waves = (N / 16) * (Nw / 64);
        gat_gemm_wmma<<<cdiv(waves, TB / 32), TB, 0, stream>>>(xb, Wb + 4 * 16384, h, N, Nw);
        gat_node_prep<<<cdiv(N * H, TB), TB, 0, stream>>>(
            h, asrcL, adstL, ssb, sdb, mxb, dnb, N, H, C);
        gat_zero_f32<<<cdiv(N * Nw, TB), TB, 0, stream>>>(acc, N * Nw);
        gat_edge_max<1><<<cdiv(E, TB), TB, 0, stream>>>(src, dst, ssb, sdb, mxb, E);
        gat_edge_expsum<1><<<cdiv(E, TB), TB, 0, stream>>>(src, dst, ssb, sdb, mxb, dnb, E);
        gat_edge_aggr<1, 64><<<cdiv(E, TB / 32), TB, 0, stream>>>(
            src, dst, ssb, sdb, mxb, dnb, h, acc, E);
        gat_finalize<<<cdiv(N * Nw, TB), TB, 0, stream>>>(
            acc, bL, nullptr, (float*)d_out, N, Nw, 0);
    }
}
